// MACE_29729763623062
// MI455X (gfx1250) — matrix operand order repaired and checked by reading, not compiled, not run on hardware
//
#include <hip/hip_runtime.h>
#include <math.h>

// ---------------- types ----------------
typedef __attribute__((ext_vector_type(16))) _Float16 v16h;
typedef __attribute__((ext_vector_type(8)))  float    v8f;

#define AVG_NN_INV (1.0f/20.0f)

__device__ const int d_LMTOL[16] = {0,1,1,1,2,2,2,2,2,3,3,3,3,3,3,3};

__device__ __forceinline__ float silu_f(float x) { return x / (1.0f + __expf(-x)); }

// ---------------- WMMA helpers ----------------
// A fragment (16x32 f16): lane L: m=L&15, h=L>>4; elems 0..7 -> K=8h+e, elems 8..15 -> K=16+8h+(e-8)
__device__ __forceinline__ v16h load_a_frag(const _Float16* base, int row_stride, int elem_stride) {
  int lane = threadIdx.x & 31;
  int m = lane & 15, h = lane >> 4;
  const _Float16* row = base + (size_t)m * row_stride;
  v16h a;
#pragma unroll
  for (int e = 0; e < 8; ++e) a[e] = row[(8*h + e) * elem_stride];
#pragma unroll
  for (int e = 0; e < 8; ++e) a[8+e] = row[(16 + 8*h + e) * elem_stride];
  return a;
}

// B fragment pre-packed: frag f = 32 lanes x 16 halves contiguous
__device__ __forceinline__ v16h load_b_frag(const _Float16* frags, int f) {
  int lane = threadIdx.x & 31;
  return *reinterpret_cast<const v16h*>(frags + ((size_t)f * 32 + lane) * 16);
}

__device__ __forceinline__ v8f wmma16(v16h a, v16h b, v8f c) {
  return __builtin_amdgcn_wmma_f32_16x16x32_f16(false, a, false, b, (short)0, c, false, false);
}

// store D tile (16x16) to LDS f16 row-major with optional silu.
// D layout: lane L: n=L&15, h=L>>4; vgpr v holds D[v+8h][n]
__device__ __forceinline__ void store_d_lds(v8f acc, _Float16* base, int row_stride, int col0, bool do_silu) {
  int lane = threadIdx.x & 31;
  int n = lane & 15, h = lane >> 4;
#pragma unroll
  for (int v = 0; v < 8; ++v) {
    float x = acc[v];
    if (do_silu) x = silu_f(x);
    base[(size_t)(v + 8*h) * row_stride + col0 + n] = (_Float16)x;
  }
}

// store D tile to LDS f32
__device__ __forceinline__ void store_d_lds_f32(v8f acc, float* base, int row_stride, int col0) {
  int lane = threadIdx.x & 31;
  int n = lane & 15, h = lane >> 4;
#pragma unroll
  for (int v = 0; v < 8; ++v)
    base[(size_t)(v + 8*h) * row_stride + col0 + n] = acc[v];
}

// ---------------- weight packing: row-major (K x Ncols) f32 -> B fragments f16 ----------------
// B layout (KxN, 32-row K-block): lane L: n=L&15, h=L>>4; elem e<8 holds B[kt*32 + 8h + e][n], e>=8 holds B[kt*32 + 16 + 8h + (e-8)][n]
__global__ void k_pack_bfrag(const float* __restrict__ W, _Float16* __restrict__ out, int K, int Ncols) {
  int f = blockIdx.x;
  int nnt = Ncols >> 4;
  int kt = f / nnt, nt = f % nnt;
  int lane = threadIdx.x;
  int n = nt*16 + (lane & 15);
  int h = lane >> 4;
  _Float16* o = out + ((size_t)f * 32 + lane) * 16;
#pragma unroll
  for (int e = 0; e < 16; ++e) {
    int k = kt*32 + (e < 8 ? 8*h + e : 16 + 8*h + (e - 8));
    float v = (k < K) ? W[(size_t)k * Ncols + n] : 0.0f;
    o[e] = (_Float16)v;
  }
}

// ---------------- species / embedding ----------------
__global__ void k_embed(const float* __restrict__ na, const float* __restrict__ E0w,
                        const float* __restrict__ emb, int* __restrict__ spec,
                        float* __restrict__ e0n, float* __restrict__ h0, int N_) {
  int i = blockIdx.x * 256 + threadIdx.x;
  if (i >= N_ * 64) return;
  int nn = i >> 6, c = i & 63;
  int sp = 0;
  const float* row = na + (size_t)nn * 10;
#pragma unroll
  for (int z = 0; z < 10; ++z) if (row[z] > 0.5f) sp = z;
  h0[i] = emb[sp * 64 + c];
  if (c == 0) { spec[nn] = sp; e0n[nn] = E0w[sp]; }
}

// ---------------- edge geometry: Y (E,16) f32 and bessel*cutoff (E,8) f16 ----------------
__global__ void k_geom(const float* __restrict__ pos, const int* __restrict__ ei,
                       _Float16* __restrict__ rb, float* __restrict__ Yg, int E_) {
  int e = blockIdx.x * 256 + threadIdx.x;
  if (e >= E_) return;
  int s = ei[e], d = ei[E_ + e];
  float x = pos[s*3+0] - pos[d*3+0];
  float y = pos[s*3+1] - pos[d*3+1];
  float z = pos[s*3+2] - pos[d*3+2];
  float r  = sqrtf(x*x + y*y + z*z);
  float rs = fmaxf(r, 1e-6f);
  float inv = 1.0f / rs;
  float ux = x*inv, uy = y*inv, uz = z*inv;
  float x2 = ux*ux, y2 = uy*uy, z2 = uz*uz;
  const float s3 = 1.7320508075688772f, s5 = 2.2360679774997896f, s15 = 3.872983346207417f;
  const float c33 = 2.091650066335189f;   // sqrt(35/8)
  const float cm2 = 10.246950765959598f;  // sqrt(105)
  const float c31 = 1.6201851746019651f;  // sqrt(21/8)
  const float c30 = 1.3228756555322954f;  // sqrt(7/4)
  const float c32 = 5.123475382979799f;   // sqrt(105/4)
  float* Y = Yg + (size_t)e * 16;
  Y[0]  = 1.0f;
  Y[1]  = s3 * ux;  Y[2] = s3 * uy;  Y[3] = s3 * uz;
  Y[4]  = s15 * ux * uy;
  Y[5]  = s15 * uy * uz;
  Y[6]  = 0.5f * s5 * (3.0f*z2 - 1.0f);
  Y[7]  = s15 * ux * uz;
  Y[8]  = 0.5f * s15 * (x2 - y2);
  Y[9]  = c33 * uy * (3.0f*x2 - y2);
  Y[10] = cm2 * ux * uy * uz;
  Y[11] = c31 * uy * (5.0f*z2 - 1.0f);
  Y[12] = c30 * uz * (5.0f*z2 - 3.0f);
  Y[13] = c31 * ux * (5.0f*z2 - 1.0f);
  Y[14] = c32 * uz * (x2 - y2);
  Y[15] = c33 * ux * (x2 - 3.0f*y2);
  // bessel * polynomial cutoff, R_MAX = 5
  float xc = rs * 0.2f;
  float env = 0.0f;
  if (xc < 1.0f) {
    float x5 = xc*xc*xc*xc*xc;
    env = 1.0f - 21.0f*x5 + 35.0f*x5*xc - 15.0f*x5*xc*xc;
  }
  const float pref = 0.6324555320336759f;  // sqrt(2/5)
  const float w = 0.6283185307179586f;     // pi/5
  _Float16* rbr = rb + (size_t)e * 8;
#pragma unroll
  for (int nb = 1; nb <= 8; ++nb) {
    float v = pref * __sinf((float)nb * w * rs) * inv * env;
    rbr[nb-1] = (_Float16)v;
  }
}

// ---------------- hu = h @ w_up  (16-node tiles) -> f16, coalesced via LDS ----------------
__global__ void __launch_bounds__(128)
k_hu(const float* __restrict__ hin, const _Float16* __restrict__ wupf,
     _Float16* __restrict__ hu, int N_) {
  __shared__ _Float16 hb[16 * 72];
  int n0 = blockIdx.x * 16;
  int tid = threadIdx.x;
  bool full = (n0 + 16 <= N_);
  for (int i = tid; i < 1024; i += 128) {
    int m = i >> 6, c = i & 63;
    float v = (full || n0 + m < N_) ? hin[(size_t)(n0 + m) * 64 + c] : 0.0f;
    hb[m * 72 + c] = (_Float16)v;
  }
  __syncthreads();
  int wid = tid >> 5;
  v16h a0 = load_a_frag(hb, 72, 1);
  v16h a1 = load_a_frag(hb + 32, 72, 1);
  v8f acc = {};
  acc = wmma16(a0, load_b_frag(wupf, 0*4 + wid), acc);
  acc = wmma16(a1, load_b_frag(wupf, 1*4 + wid), acc);
  __syncthreads();                       // all waves consumed hb
  store_d_lds(acc, hb, 72, wid * 16, false);
  __syncthreads();
  if (full) {
    for (int i = tid; i < 1024; i += 128)
      hu[(size_t)n0 * 64 + i] = hb[(i >> 6) * 72 + (i & 63)];
  } else {
    for (int i = tid; i < 1024; i += 128) {
      int m = i >> 6;
      if (n0 + m < N_) hu[(size_t)n0 * 64 + i] = hb[m * 72 + (i & 63)];
    }
  }
}

// ---------------- fused radial-MLP + message + scatter ----------------
// 8 waves/block, 16 edges per wave.
__global__ void __launch_bounds__(256)
k_msg(const _Float16* __restrict__ rb, const float* __restrict__ Yg,
      const _Float16* __restrict__ hu, const int* __restrict__ ei,
      const _Float16* __restrict__ w0f, const _Float16* __restrict__ w1f,
      const _Float16* __restrict__ w2f, const _Float16* __restrict__ w3f,
      float* __restrict__ Abuf, int E_) {
  __shared__ _Float16 xs[8][16 * 72];
  __shared__ _Float16 ys[8][16 * 72];
  __shared__ float    Ys[8][256];
  __shared__ int      dsts[8][16];
  __shared__ int      srcs[8][16];

  int tid = threadIdx.x;
  int wid = tid >> 5, lane = tid & 31;
  int e0 = (blockIdx.x * 8 + wid) * 16;

  // warm L0/L2 with the big packed W3 (32KB) once per block
  if (wid == 0) {
    const char* wp = (const char*)w3f;
    for (int i = lane; i < 512; i += 32) __builtin_prefetch(wp + (size_t)i * 64, 0, 3);
  }

  // stage rb into cols 0..7, zero 8..31
  for (int i = lane; i < 16 * 32; i += 32) {
    int m = i >> 5, c = i & 31;
    int e = e0 + m;
    float v = 0.0f;
    if (c < 8 && e < E_) v = (float)rb[(size_t)e * 8 + c];
    xs[wid][m * 72 + c] = (_Float16)v;
  }
  for (int i = lane; i < 256; i += 32) {
    int m = i >> 4, k = i & 15;
    int e = e0 + m;
    Ys[wid][i] = (e < E_) ? Yg[(size_t)e * 16 + k] : 0.0f;
  }
  if (lane < 16) {
    int e = e0 + lane;
    dsts[wid][lane] = (e < E_) ? ei[E_ + e] : 0;
    srcs[wid][lane] = (e < E_) ? ei[e] : 0;
  }
  __syncthreads();

  // MLP layer 1 : (16x32) @ (32x64), K really 8 (padded with zero weights/inputs)
  {
    v16h a0 = load_a_frag(xs[wid], 72, 1);
    for (int nt = 0; nt < 4; ++nt) {
      v8f acc = {};
      acc = wmma16(a0, load_b_frag(w0f, nt), acc);
      store_d_lds(acc, ys[wid], 72, nt*16, true);
    }
  }
  __syncthreads();
  // MLP layer 2
  {
    v16h a0 = load_a_frag(ys[wid], 72, 1);
    v16h a1 = load_a_frag(ys[wid] + 32, 72, 1);
    for (int nt = 0; nt < 4; ++nt) {
      v8f acc = {};
      acc = wmma16(a0, load_b_frag(w1f, 0*4 + nt), acc);
      acc = wmma16(a1, load_b_frag(w1f, 1*4 + nt), acc);
      store_d_lds(acc, xs[wid], 72, nt*16, true);
    }
  }
  __syncthreads();
  // MLP layer 3
  {
    v16h a0 = load_a_frag(xs[wid], 72, 1);
    v16h a1 = load_a_frag(xs[wid] + 32, 72, 1);
    for (int nt = 0; nt < 4; ++nt) {
      v8f acc = {};
      acc = wmma16(a0, load_b_frag(w2f, 0*4 + nt), acc);
      acc = wmma16(a1, load_b_frag(w2f, 1*4 + nt), acc);
      store_d_lds(acc, ys[wid], 72, nt*16, true);
    }
  }
  __syncthreads();
  // stage hu[src] rows into xs
  for (int i = lane; i < 16 * 64; i += 32) {
    int m = i >> 6, c = i & 63;
    xs[wid][m * 72 + c] = hu[(size_t)srcs[wid][m] * 64 + c];
  }
  __syncthreads();

  // MLP layer 4 (-> rw, 256 cols in 16 tiles) fused with message scatter
  v16h a0 = load_a_frag(ys[wid], 72, 1);
  v16h a1 = load_a_frag(ys[wid] + 32, 72, 1);
  int n = lane & 15, h = lane >> 4;
  const int kst[4] = {0, 1, 4, 9};
  const int kct[4] = {1, 3, 5, 7};
  for (int t = 0; t < 16; ++t) {
    v8f acc = {};
    acc = wmma16(a0, load_b_frag(w3f, 0*16 + t), acc);
    acc = wmma16(a1, load_b_frag(w3f, 1*16 + t), acc);
    int j = t*16 + n;          // rw column = 4*c + l
    int c = j >> 2, l = j & 3;
    int ks = kst[l], kc = kct[l];
#pragma unroll
    for (int v = 0; v < 8; ++v) {
      int m = v + 8*h;
      float base = acc[v] * (float)xs[wid][m * 72 + c];
      float* Arow = Abuf + (size_t)dsts[wid][m] * 1024 + c * 16;
      const float* Yrow = &Ys[wid][m * 16];
      for (int kk = 0; kk < kc; ++kk) {
        int k = ks + kk;
        atomicAdd(Arow + k, base * Yrow[k]);
      }
    }
  }
}

// ---------------- A-linear: Alin[n,d,k] = sum_c (A[n,c,k]/20) * w_lin[lm(k)][c,d] ----------------
// LDS holds the tile k-sliced: At[k] is a 16x64 (stride 72) row-major slice, so
// A-fragment loads are contiguous (vectorize to ds_load_b128). D is written back
// in-place into the owning k-slice, then copied out coalesced in global layout.
__global__ void __launch_bounds__(128)
k_lin(const float* __restrict__ A, const _Float16* __restrict__ wlinf,
      _Float16* __restrict__ Alin, int N_) {
  __shared__ _Float16 At[16][16 * 72];
  int n0 = blockIdx.x * 16;
  int tid = threadIdx.x;
  bool full = (n0 + 16 <= N_);
  for (int i = tid; i < 16 * 1024; i += 128) {
    int m = i >> 10, r = i & 1023;
    int c = r >> 4, k = r & 15;
    float v = (full || n0 + m < N_) ? A[(size_t)(n0 + m) * 1024 + r] * AVG_NN_INV : 0.0f;
    At[k][m * 72 + c] = (_Float16)v;
  }
  __syncthreads();
  int wid = tid >> 5, lane = tid & 31;
  int n = lane & 15, h = lane >> 4;
  for (int kk = 0; kk < 4; ++kk) {
    int k = wid * 4 + kk;
    int lm = d_LMTOL[k];
    const _Float16* wf = wlinf + (size_t)lm * 8 * 512;
    v16h a0 = load_a_frag(At[k], 72, 1);       // consumed into regs before writes below
    v16h a1 = load_a_frag(At[k] + 32, 72, 1);
    for (int nt = 0; nt < 4; ++nt) {
      v8f acc = {};
      acc = wmma16(a0, load_b_frag(wf, 0*4 + nt), acc);
      acc = wmma16(a1, load_b_frag(wf, 1*4 + nt), acc);
#pragma unroll
      for (int v = 0; v < 8; ++v) {
        int m = v + 8*h;
        At[k][m * 72 + nt*16 + n] = (_Float16)acc[v];   // own k-slice only
      }
    }
  }
  __syncthreads();
  if (full) {
    for (int i = tid; i < 16 * 1024; i += 128) {
      int m = i >> 10, r = i & 1023;
      Alin[(size_t)n0 * 1024 + i] = At[r & 15][m * 72 + (r >> 4)];
    }
  } else {
    for (int i = tid; i < 16 * 1024; i += 128) {
      int m = i >> 10, r = i & 1023;
      if (n0 + m < N_) Alin[(size_t)n0 * 1024 + i] = At[r & 15][m * 72 + (r >> 4)];
    }
  }
}

// ---------------- B features + species contraction + @w_prod (+optional skip add) ----------------
__global__ void __launch_bounds__(128)
k_prod(const _Float16* __restrict__ Alin, const float* __restrict__ wcon,
       const int* __restrict__ spec, const _Float16* __restrict__ wprodf,
       const float* __restrict__ add, float* __restrict__ hout, int N_) {
  __shared__ _Float16 mb[16 * 72];
  __shared__ float    ob[16 * 68];
  int n0 = blockIdx.x * 16;
  int tid = threadIdx.x;
  bool full = (n0 + 16 <= N_);
  for (int i = tid; i < 1024; i += 128) {
    int m = i >> 6, c = i & 63;
    float mval = 0.0f;
    if (full || n0 + m < N_) {
      const _Float16* ar = Alin + (size_t)(n0 + m) * 1024 + c * 16;
      float a0 = (float)ar[0];
      float s0 = a0 * a0;
      float s1 = 0.0f, s2 = 0.0f, s3v = 0.0f;
#pragma unroll
      for (int k = 1; k < 4; ++k)  { float v = (float)ar[k]; s1 += v*v; }
#pragma unroll
      for (int k = 4; k < 9; ++k)  { float v = (float)ar[k]; s2 += v*v; }
#pragma unroll
      for (int k = 9; k < 16; ++k) { float v = (float)ar[k]; s3v += v*v; }
      const float* wc = wcon + (size_t)spec[n0 + m] * 5 * 64 + c;
      mval = a0*wc[0] + s0*wc[64] + s1*wc[128] + s2*wc[192] + s3v*wc[256];
    }
    mb[m * 72 + c] = (_Float16)mval;
  }
  __syncthreads();
  int wid = tid >> 5;
  v16h a0 = load_a_frag(mb, 72, 1);
  v16h a1 = load_a_frag(mb + 32, 72, 1);
  v8f acc = {};
  acc = wmma16(a0, load_b_frag(wprodf, 0*4 + wid), acc);
  acc = wmma16(a1, load_b_frag(wprodf, 1*4 + wid), acc);
  store_d_lds_f32(acc, ob, 68, wid * 16);
  __syncthreads();
  for (int i = tid; i < 1024; i += 128) {
    int m = i >> 6, c = i & 63;
    if (full || n0 + m < N_) {
      float o = ob[m * 68 + c];
      if (add) o += add[(size_t)(n0 + m) * 64 + c];
      hout[(size_t)(n0 + m) * 64 + c] = o;
    }
  }
}

// ---------------- species-conditioned skip: sc[n,d] = sum_c h1[n,c]*w_sc[spec,c,d] ----------------
__global__ void k_sc(const float* __restrict__ h1, const float* __restrict__ wsc,
                     const int* __restrict__ spec, float* __restrict__ sc, int N_) {
  int nn = blockIdx.x;
  int d = threadIdx.x;  // 64
  const float* W = wsc + (size_t)spec[nn] * 4096;
  const float* hr = h1 + (size_t)nn * 64;
  float acc = 0.0f;
  for (int c = 0; c < 64; ++c) acc += hr[c] * W[c * 64 + d];
  sc[(size_t)nn * 64 + d] = acc;
}

// ---------------- readouts + graph reduction ----------------
__global__ void k_energy(const float* __restrict__ h1, const float* __restrict__ h2,
                         const float* __restrict__ wr1, const float* __restrict__ wa,
                         const float* __restrict__ wb, const float* __restrict__ e0n,
                         const int* __restrict__ batch, float* __restrict__ out, int N_) {
  __shared__ float red[64];
  __shared__ float hid[16];
  int nn = blockIdx.x;
  int t = threadIdx.x;  // 64
  red[t] = h1[(size_t)nn * 64 + t] * wr1[t];
  __syncthreads();
  for (int s = 32; s > 0; s >>= 1) {
    if (t < s) red[t] += red[t + s];
    __syncthreads();
  }
  if (t < 16) {
    float a = 0.0f;
    for (int c = 0; c < 64; ++c) a += h2[(size_t)nn * 64 + c] * wa[c * 16 + t];
    hid[t] = silu_f(a);
  }
  __syncthreads();
  if (t == 0) {
    float e2 = 0.0f;
#pragma unroll
    for (int k = 0; k < 16; ++k) e2 += hid[k] * wb[k];
    atomicAdd(&out[batch[nn]], e0n[nn] + red[0] + e2);
  }
}

// ================= host launcher =================
extern "C" void kernel_launch(void* const* d_in, const int* in_sizes, int n_in,
                              void* d_out, int out_size, void* d_ws, size_t ws_size,
                              hipStream_t stream) {
  // ---- locate inputs by element count (all top-level leaves are unique) ----
  auto find = [&](int sz) -> int {
    for (int i = 0; i < n_in; ++i) if (in_sizes[i] == sz) return i;
    return -1;
  };
  int i_pos = find(30000), i_na = find(100000), i_E0 = find(10), i_emb = find(640);
  int i_wsc = find(40960), i_wr1 = find(64), i_wa = find(1024), i_wb = find(16);
  int i_ei = find(200000), i_batch = find(10000);
  int l0 = -1, l1 = -1;
  for (int i = 0; i < n_in; ++i) {
    if (in_sizes[i] == 512) { if (l0 < 0) l0 = i; else if (l1 < 0) l1 = i; }
  }
  if (i_pos < 0 || i_na < 0 || i_E0 < 0 || i_emb < 0 || i_wsc < 0 || i_wr1 < 0 ||
      i_wa < 0 || i_wb < 0 || i_ei < 0 || i_batch < 0 || l0 < 0 || l1 < 0) return;

  struct LayerIdx { int r0, r1, r2, r3, wup, wlin, wcon, wprod; };
  auto layer_idx = [&](int s) -> LayerIdx {
    LayerIdx L;
    L.r0 = s; L.r1 = s + 1; L.r2 = s + 2; L.r3 = s + 3;
    if (in_sizes[s + 4] == 4096) {           // insertion order: w_up, w_lin, w_con, w_prod
      L.wup = s + 4; L.wlin = s + 5; L.wcon = s + 6; L.wprod = s + 7;
    } else {                                  // sorted keys: w_con, w_lin, w_prod, w_up
      L.wcon = s + 4; L.wlin = s + 5; L.wprod = s + 6; L.wup = s + 7;
    }
    return L;
  };
  LayerIdx L1 = layer_idx(l0), L2 = layer_idx(l1);

  const int N_ = in_sizes[i_batch];
  const int E_ = in_sizes[i_ei] / 2;

  const float* pos = (const float*)d_in[i_pos];
  const float* na  = (const float*)d_in[i_na];
  const float* E0w = (const float*)d_in[i_E0];
  const float* emb = (const float*)d_in[i_emb];
  const float* wsc = (const float*)d_in[i_wsc];
  const float* wr1 = (const float*)d_in[i_wr1];
  const float* wa  = (const float*)d_in[i_wa];
  const float* wb  = (const float*)d_in[i_wb];
  const int* ei    = (const int*)d_in[i_ei];
  const int* batch = (const int*)d_in[i_batch];

  // ---- carve workspace ----
  char* p = (char*)d_ws;
  auto carve = [&](size_t bytes) -> void* {
    void* r = (void*)p;
    p += (bytes + 255) & ~(size_t)255;
    return r;
  };
  _Float16* rb   = (_Float16*)carve((size_t)E_ * 8 * 2);
  float*    Yg   = (float*)   carve((size_t)E_ * 16 * 4);
  int*      spec = (int*)     carve((size_t)N_ * 4);
  float*    e0n  = (float*)   carve((size_t)N_ * 4);
  float*    h0   = (float*)   carve((size_t)N_ * 64 * 4);
  float*    h1   = (float*)   carve((size_t)N_ * 64 * 4);
  float*    h2   = (float*)   carve((size_t)N_ * 64 * 4);
  float*    sc   = (float*)   carve((size_t)N_ * 64 * 4);
  _Float16* hu   = (_Float16*)carve((size_t)N_ * 64 * 2);
  float*    A    = (float*)   carve((size_t)N_ * 1024 * 4);
  _Float16* Alin = (_Float16*)carve((size_t)N_ * 1024 * 2);
  // packed weight fragments (f16), per layer: w0(2048) w1(4096) w2(4096) w3(16384) wup(4096) wlin(16384) wprod(4096)
  _Float16* packA = (_Float16*)carve((size_t)51200 * 2);
  _Float16* packB = (_Float16*)carve((size_t)51200 * 2);

  struct Pk { _Float16 *w0, *w1, *w2, *w3, *wup, *wlin, *wprod; };
  auto mkpk = [](_Float16* b) -> Pk {
    Pk k;
    k.w0 = b; k.w1 = b + 2048; k.w2 = b + 6144; k.w3 = b + 10240;
    k.wup = b + 26624; k.wlin = b + 30720; k.wprod = b + 47104;
    return k;
  };
  Pk P1 = mkpk(packA), P2 = mkpk(packB);

  // ---- pack weights into B fragments ----
  auto pack_layer = [&](const LayerIdx& L, const Pk& P) {
    k_pack_bfrag<<<4, 32, 0, stream>>>((const float*)d_in[L.r0], P.w0, 8, 64);      // 8x64 padded to K=32
    k_pack_bfrag<<<8, 32, 0, stream>>>((const float*)d_in[L.r1], P.w1, 64, 64);
    k_pack_bfrag<<<8, 32, 0, stream>>>((const float*)d_in[L.r2], P.w2, 64, 64);
    k_pack_bfrag<<<32, 32, 0, stream>>>((const float*)d_in[L.r3], P.w3, 64, 256);
    k_pack_bfrag<<<8, 32, 0, stream>>>((const float*)d_in[L.wup], P.wup, 64, 64);
    for (int l = 0; l < 4; ++l)
      k_pack_bfrag<<<8, 32, 0, stream>>>(((const float*)d_in[L.wlin]) + (size_t)l * 4096,
                                         P.wlin + (size_t)l * 4096, 64, 64);
    k_pack_bfrag<<<8, 32, 0, stream>>>((const float*)d_in[L.wprod], P.wprod, 64, 64);
  };
  pack_layer(L1, P1);
  pack_layer(L2, P2);

  // ---- pipeline ----
  hipMemsetAsync(d_out, 0, (size_t)out_size * 4, stream);
  hipMemsetAsync(A, 0, (size_t)N_ * 1024 * 4, stream);

  k_embed<<<(N_ * 64 + 255) / 256, 256, 0, stream>>>(na, E0w, emb, spec, e0n, h0, N_);
  k_geom<<<(E_ + 255) / 256, 256, 0, stream>>>(pos, ei, rb, Yg, E_);

  // layer 1
  k_hu<<<(N_ + 15) / 16, 128, 0, stream>>>(h0, P1.wup, hu, N_);
  k_msg<<<(E_ + 127) / 128, 256, 0, stream>>>(rb, Yg, hu, ei, P1.w0, P1.w1, P1.w2, P1.w3, A, E_);
  k_lin<<<(N_ + 15) / 16, 128, 0, stream>>>(A, P1.wlin, Alin, N_);
  k_prod<<<(N_ + 15) / 16, 128, 0, stream>>>(Alin, (const float*)d_in[L1.wcon], spec, P1.wprod,
                                             nullptr, h1, N_);
  k_sc<<<N_, 64, 0, stream>>>(h1, wsc, spec, sc, N_);

  // layer 2
  hipMemsetAsync(A, 0, (size_t)N_ * 1024 * 4, stream);
  k_hu<<<(N_ + 15) / 16, 128, 0, stream>>>(h1, P2.wup, hu, N_);
  k_msg<<<(E_ + 127) / 128, 256, 0, stream>>>(rb, Yg, hu, ei, P2.w0, P2.w1, P2.w2, P2.w3, A, E_);
  k_lin<<<(N_ + 15) / 16, 128, 0, stream>>>(A, P2.wlin, Alin, N_);
  k_prod<<<(N_ + 15) / 16, 128, 0, stream>>>(Alin, (const float*)d_in[L2.wcon], spec, P2.wprod,
                                             sc, h2, N_);

  k_energy<<<N_, 64, 0, stream>>>(h1, h2, wr1, wa, wb, e0n, batch, (float*)d_out, N_);
}
